// BertSelfAttention_24421184045274
// MI455X (gfx1250) — matrix operand order repaired and checked by reading, not compiled, not run on hardware
//
#include <hip/hip_runtime.h>
#include <hip/hip_bf16.h>

// ---------------------------------------------------------------------------
// Dual-stream BERT self-attention for MI455X (gfx1250), bf16 WMMA pipeline.
//   Stage 1: fp32 -> bf16 conversion of activations + weights
//   Stage 2: 6 projection GEMMs (x @ W^T + b) via v_wmma_f32_16x16x32_bf16
//            K/Ko stored [bh][s][dh], V/Vo stored transposed [bh][dh][s]
//   Stage 3: flash attention, one wave per (b,h,16-query) tile, online softmax
//            over the 2048 concatenated keys, all matmuls via WMMA.
// ---------------------------------------------------------------------------

typedef __bf16 v16bf __attribute__((ext_vector_type(16)));
typedef __bf16 v8bf  __attribute__((ext_vector_type(8)));
typedef float  v8f   __attribute__((ext_vector_type(8)));

#define WMMA_BF16(A, B, C) \
  __builtin_amdgcn_wmma_f32_16x16x32_bf16(false, (A), false, (B), (short)0, (C), false, false)

static __device__ __forceinline__ __bf16 f2bf(float f) {
  unsigned u = __builtin_bit_cast(unsigned, f);
  u += 0x7FFFu + ((u >> 16) & 1u);               // round-to-nearest-even
  unsigned short h = (unsigned short)(u >> 16);
  return __builtin_bit_cast(__bf16, h);
}

// Build a 16x32 A-fragment slice (WMMA-K = k0..k0+31) from a row-major row.
// Lane layout: lanes 0-15 hold K {0..15}; lanes 16-31 hold K {16..31}.
static __device__ __forceinline__ v16bf load_afrag(const __bf16* row, int k0, bool hi) {
  v8bf a0 = *(const v8bf*)(row + k0 + (hi ? 16 : 0));
  v8bf a1 = *(const v8bf*)(row + k0 + 8 + (hi ? 16 : 0));
  return __builtin_shufflevector(a0, a1, 0, 1, 2, 3, 4, 5, 6, 7,
                                 8, 9, 10, 11, 12, 13, 14, 15);
}

// ---------------------------------------------------------------------------
__global__ void cvt_bf16_kernel(const float* __restrict__ src,
                                __bf16* __restrict__ dst, int n) {
  int i = blockIdx.x * blockDim.x + threadIdx.x;
  int stride = gridDim.x * blockDim.x;
  for (; i < n; i += stride) dst[i] = f2bf(src[i]);
}

// ---------------------------------------------------------------------------
// Y = X @ W^T + b.  X:[4096,768] bf16 row-major, W:[768,768] bf16 row-major
// (nn.Linear convention, W[out][in]), bias fp32.
// mode 0: out[(b*12+h)*1024 + s][64]   (split heads, row-major, for Q/K/Qo/Ko)
// mode 1: out[((b*12+h)*64 + dh)][1024] (split heads, dh-major,  for V/Vo)
__global__ void __launch_bounds__(256)
proj_gemm_kernel(const __bf16* __restrict__ X, const __bf16* __restrict__ W,
                 const float* __restrict__ bias, __bf16* __restrict__ out, int mode) {
  const int lane  = threadIdx.x & 31;
  const int lanei = lane & 15;
  const bool hi   = lane >= 16;
  const int gw = blockIdx.x * (blockDim.x >> 5) + (threadIdx.x >> 5);
  const int mt = gw / 48;   // 256 row tiles of 16 (B*S = 4096)
  const int nt = gw % 48;   // 48 col tiles of 16 (D = 768)

  const __bf16* arow = X + (size_t)(mt * 16 + lanei) * 768;
  // B operand: element (k,n) = W[n][k]; lane holds column n=lane%16,
  // K half selected by lane group -> one contiguous 32B load per step.
  const __bf16* brow = W + (size_t)(nt * 16 + lanei) * 768 + (hi ? 16 : 0);

  v8f c = {};
#pragma unroll
  for (int k0 = 0; k0 < 768; k0 += 32) {
    v16bf a = load_afrag(arow, k0, hi);
    v16bf b = *(const v16bf*)(brow + k0);
    c = WMMA_BF16(a, b, c);
  }

  const int n  = nt * 16 + lanei;
  const int hh = n >> 6, dh = n & 63;
  const float bv = bias[n];

  if (mode == 0) {
#pragma unroll
    for (int r = 0; r < 8; ++r) {
      int m = mt * 16 + r + (hi ? 8 : 0);
      int bb = m >> 10, s = m & 1023;
      out[(size_t)((bb * 12 + hh) * 1024 + s) * 64 + dh] = f2bf(c[r] + bv);
    }
  } else {
    int m0 = mt * 16 + (hi ? 8 : 0);
    int bb = m0 >> 10, s0 = m0 & 1023;
    v8bf v;
#pragma unroll
    for (int r = 0; r < 8; ++r) v[r] = f2bf(c[r] + bv);
    *(v8bf*)(out + (size_t)((bb * 12 + hh) * 64 + dh) * 1024 + s0) = v;  // 16B store
  }
}

// ---------------------------------------------------------------------------
// Flash attention over 2048 concatenated keys (self stream uses Q/K + mask,
// other stream uses Qo/Ko, shared online softmax).  One wave per 16-query tile.
__global__ void __launch_bounds__(256)
attn_kernel(const __bf16* __restrict__ Qbf, const __bf16* __restrict__ Kbf,
            const __bf16* __restrict__ Qobf, const __bf16* __restrict__ Kobf,
            const __bf16* __restrict__ Vt, const __bf16* __restrict__ Vot,
            const float* __restrict__ mask, float* __restrict__ outp) {
  __shared__ unsigned short plds[8][16 * 40];  // per-wave 16x32 P tile, padded rows

  const int lane  = threadIdx.x & 31;
  const int lanei = lane & 15;
  const bool hi   = lane >= 16;
  const int wid   = threadIdx.x >> 5;
  const int gw = blockIdx.x * 8 + wid;    // 0..3071
  const int bh = gw >> 6;                 // 0..47  (b*12 + h)
  const int qt = gw & 63;                 // query tile
  const int bIdx = bh / 12, h = bh % 12;

  const __bf16* qrow  = Qbf  + (size_t)(bh * 1024 + qt * 16 + lanei) * 64;
  const __bf16* qorow = Qobf + (size_t)(bh * 1024 + qt * 16 + lanei) * 64;
  const v16bf qa0  = load_afrag(qrow, 0, hi),  qa1  = load_afrag(qrow, 32, hi);
  const v16bf qoa0 = load_afrag(qorow, 0, hi), qoa1 = load_afrag(qorow, 32, hi);

  float mrow[8], lrow[8];
  v8f o[4] = {};
#pragma unroll
  for (int r = 0; r < 8; ++r) { mrow[r] = -__builtin_inff(); lrow[r] = 0.0f; }

  for (int kb = 0; kb < 2048; kb += 32) {
    const bool self = kb < 1024;
    const __bf16* Kb = self ? Kbf : Kobf;
    const __bf16* Vb = self ? Vt : Vot;
    const int kk = self ? kb : (kb - 1024);
    const v16bf a0 = self ? qa0 : qoa0;
    const v16bf a1 = self ? qa1 : qoa1;

    // ---- scores for 32 keys: two 16x16 WMMA tiles, K(dh)=64 in two steps ----
    v8f s[2];
#pragma unroll
    for (int cg = 0; cg < 2; ++cg) {
      const __bf16* krow =
          Kb + (size_t)(bh * 1024 + kk + cg * 16 + lanei) * 64 + (hi ? 16 : 0);
      v8f acc = {};
      acc = WMMA_BF16(a0, *(const v16bf*)(krow), acc);
      acc = WMMA_BF16(a1, *(const v16bf*)(krow + 32), acc);
      const float mv = self ? mask[bIdx * 1024 + kk + cg * 16 + lanei] : 0.0f;
#pragma unroll
      for (int r = 0; r < 8; ++r) acc[r] = acc[r] * 0.125f + mv;  // 1/sqrt(64)
      s[cg] = acc;
    }

    // ---- online softmax: row stats via width-16 xor shuffles ----
    float corr[8];
#pragma unroll
    for (int r = 0; r < 8; ++r) {
      float v = fmaxf(s[0][r], s[1][r]);
#pragma unroll
      for (int x = 1; x < 16; x <<= 1) v = fmaxf(v, __shfl_xor(v, x, 32));
      const float mn = fmaxf(mrow[r], v);
      corr[r] = __expf(mrow[r] - mn);
      mrow[r] = mn;
      const float p0 = __expf(s[0][r] - mn);
      const float p1 = __expf(s[1][r] - mn);
      s[0][r] = p0; s[1][r] = p1;
      float rs = p0 + p1;
#pragma unroll
      for (int x = 1; x < 16; x <<= 1) rs += __shfl_xor(rs, x, 32);
      lrow[r] = lrow[r] * corr[r] + rs;
    }

    // ---- P (C-layout) -> LDS -> A-fragment ----
    unsigned short* pw = &plds[wid][0];
#pragma unroll
    for (int r = 0; r < 8; ++r) {
      const int row = r + (hi ? 8 : 0);
      pw[row * 40 + lanei]      = __builtin_bit_cast(unsigned short, f2bf(s[0][r]));
      pw[row * 40 + 16 + lanei] = __builtin_bit_cast(unsigned short, f2bf(s[1][r]));
    }
    asm volatile("" ::: "memory");  // keep DS store->load in program order
    const unsigned short* pr = &plds[wid][lanei * 40 + (hi ? 16 : 0)];
    const v8bf p0 = *(const v8bf*)(pr);
    const v8bf p1 = *(const v8bf*)(pr + 8);
    const v16bf pa = __builtin_shufflevector(p0, p1, 0, 1, 2, 3, 4, 5, 6, 7,
                                             8, 9, 10, 11, 12, 13, 14, 15);
    asm volatile("" ::: "memory");

    // ---- O = O*corr + P @ V  (V transposed: contiguous 32B B-fragment loads) ----
#pragma unroll
    for (int d = 0; d < 4; ++d) {
#pragma unroll
      for (int r = 0; r < 8; ++r) o[d][r] *= corr[r];
      const v16bf vb = *(const v16bf*)(
          Vb + (size_t)(bh * 64 + d * 16 + lanei) * 1024 + kk + (hi ? 16 : 0));
      o[d] = WMMA_BF16(pa, vb, o[d]);
    }
  }

  // ---- epilogue: normalize, merge heads, fp32 output [B,S,768] ----
#pragma unroll
  for (int r = 0; r < 8; ++r) lrow[r] = 1.0f / lrow[r];
#pragma unroll
  for (int d = 0; d < 4; ++d) {
#pragma unroll
    for (int r = 0; r < 8; ++r) {
      const int sIdx = qt * 16 + r + (hi ? 8 : 0);
      outp[(size_t)(bIdx * 1024 + sIdx) * 768 + h * 64 + d * 16 + lanei] =
          o[d][r] * lrow[r];
    }
  }
}

// ---------------------------------------------------------------------------
extern "C" void kernel_launch(void* const* d_in, const int* in_sizes, int n_in,
                              void* d_out, int out_size, void* d_ws, size_t ws_size,
                              hipStream_t stream) {
  (void)in_sizes; (void)n_in; (void)out_size; (void)ws_size;

  const float* hs   = (const float*)d_in[0];   // hidden_states        [4,1024,768]
  const float* hso  = (const float*)d_in[1];   // hidden_states_other  [4,1024,768]
  const float* mask = (const float*)d_in[2];   // attention_mask       [4,1,1,1024]
  const float* W[6]  = {(const float*)d_in[3],  (const float*)d_in[5],
                        (const float*)d_in[7],  (const float*)d_in[9],
                        (const float*)d_in[11], (const float*)d_in[13]};
  const float* Bv[6] = {(const float*)d_in[4],  (const float*)d_in[6],
                        (const float*)d_in[8],  (const float*)d_in[10],
                        (const float*)d_in[12], (const float*)d_in[14]};
  float* out = (float*)d_out;

  // Workspace layout (bf16 elements); total ~57.4 MB.
  constexpr size_t NX = 4096 * 768;   // activations
  constexpr size_t NW = 768 * 768;    // one weight matrix
  constexpr size_t NP = 48 * 1024 * 64;  // one projection output
  __bf16* ws   = (__bf16*)d_ws;
  __bf16* Xbf  = ws;
  __bf16* Xobf = Xbf + NX;
  __bf16* Wbf  = Xobf + NX;       // 6 consecutive weight matrices
  __bf16* Qbf  = Wbf + 6 * NW;
  __bf16* Kbf  = Qbf + NP;
  __bf16* Qobf = Kbf + NP;
  __bf16* Kobf = Qobf + NP;
  __bf16* Vt   = Kobf + NP;       // transposed [bh][dh][s]
  __bf16* Vot  = Vt + NP;

  cvt_bf16_kernel<<<512, 256, 0, stream>>>(hs, Xbf, (int)NX);
  cvt_bf16_kernel<<<512, 256, 0, stream>>>(hso, Xobf, (int)NX);
  for (int j = 0; j < 6; ++j)
    cvt_bf16_kernel<<<288, 256, 0, stream>>>(W[j], Wbf + (size_t)j * NW, (int)NW);

  // 256x48 = 12288 output tiles, 8 waves/block -> 1536 blocks.
  const dim3 pg(1536), pb(256);
  proj_gemm_kernel<<<pg, pb, 0, stream>>>(Xbf,  Wbf + 0 * NW, Bv[0], Qbf,  0); // q
  proj_gemm_kernel<<<pg, pb, 0, stream>>>(Xbf,  Wbf + 1 * NW, Bv[1], Kbf,  0); // k
  proj_gemm_kernel<<<pg, pb, 0, stream>>>(Xbf,  Wbf + 2 * NW, Bv[2], Vt,   1); // v (T)
  proj_gemm_kernel<<<pg, pb, 0, stream>>>(Xbf,  Wbf + 3 * NW, Bv[3], Qobf, 0); // qo (same stream)
  proj_gemm_kernel<<<pg, pb, 0, stream>>>(Xobf, Wbf + 4 * NW, Bv[4], Kobf, 0); // ko (other)
  proj_gemm_kernel<<<pg, pb, 0, stream>>>(Xobf, Wbf + 5 * NW, Bv[5], Vot,  1); // vo (other, T)

  // 48 bh * 64 query tiles = 3072 waves, 8 waves/block -> 384 blocks.
  attn_kernel<<<384, 256, 0, stream>>>(Qbf, Kbf, Qobf, Kobf, Vt, Vot, mask, out);
}
